// BertAttention_32684701122651
// MI455X (gfx1250) — compile-verified, matrix operand order checked by reading, not run on hardware
//
#include <hip/hip_runtime.h>
#include <math.h>

// ---------------------------------------------------------------------------
// Types for gfx1250 WMMA (wave32): v_wmma_f32_16x16x32_bf16
// ---------------------------------------------------------------------------
typedef __bf16 bf16_t;
typedef __attribute__((ext_vector_type(16))) __bf16 v16bf;
typedef __attribute__((ext_vector_type(8)))  __bf16 bf16x8;
typedef __attribute__((ext_vector_type(4)))  __bf16 bf16x4;
typedef __attribute__((ext_vector_type(8)))  float  v8f;
typedef __attribute__((ext_vector_type(4)))  float  f32x4;

// gfx1250 async global->LDS copy path (ASYNCcnt), guarded so the build can't
// regress if this toolchain doesn't declare the builtins. ROCm 7.2 clang-22
// declares the b128 variant as:
//   void __builtin_amdgcn_global_load_async_to_lds_b128(
//       int4 AS(1)* gsrc, int4 AS(3)* ldst, imm int offset, imm int cpol)
#if defined(__has_builtin)
#  if __has_builtin(__builtin_amdgcn_global_load_async_to_lds_b128) && \
      __has_builtin(__builtin_amdgcn_s_wait_asynccnt)
#    define USE_ASYNC_LDS 1
#  else
#    define USE_ASYNC_LDS 0
#  endif
#else
#  define USE_ASYNC_LDS 0
#endif

__device__ __forceinline__ void stage_b128(bf16_t* lds_dst, const bf16_t* g_src) {
#if USE_ASYNC_LDS
  typedef int i32x4v __attribute__((vector_size(16)));
  __builtin_amdgcn_global_load_async_to_lds_b128(
      (__attribute__((address_space(1))) i32x4v*)g_src,
      (__attribute__((address_space(3))) i32x4v*)lds_dst,
      0, 0);
#else
  *(bf16x8*)lds_dst = *(const bf16x8*)g_src;
#endif
}
__device__ __forceinline__ void stage_wait() {
#if USE_ASYNC_LDS
  __builtin_amdgcn_s_wait_asynccnt(0);
#endif
}

__device__ __forceinline__ bf16_t f2bf(float f) {
  unsigned int u = __builtin_bit_cast(unsigned int, f);
  unsigned int r = (u + 0x7FFFu + ((u >> 16) & 1u)) >> 16;  // RNE
  unsigned short s = (unsigned short)r;
  return __builtin_bit_cast(bf16_t, s);
}

// ---------------------------------------------------------------------------
// Fragment loader. ISA 7.12.2: for 16-bit 16x32 operands, lane (r = lane&15,
// hf = lane>>4) holds K = hf*8 + {0..7} and hf*8+16 + {0..7}: two contiguous
// 16-byte runs when the operand is stored row-per-lane ([m][k] for A,
// [n][k] for B). -> 2x b128 loads per fragment instead of 16x u16.
// Works for LDS and global pointers alike.
// ---------------------------------------------------------------------------
__device__ __forceinline__ v16bf frag_row(const bf16_t* p, int ld) {
  int lane = threadIdx.x & 31;
  int hf = lane >> 4, r = lane & 15;
  const bf16_t* q = p + r * ld + hf * 8;
  bf16x8 lo = *(const bf16x8*)(q);
  bf16x8 hi = *(const bf16x8*)(q + 16);
  return __builtin_shufflevector(lo, hi, 0, 1, 2, 3, 4, 5, 6, 7,
                                 8, 9, 10, 11, 12, 13, 14, 15);
}

__device__ __forceinline__ v8f wmma_bf16(v16bf a, v16bf b, v8f c) {
  return __builtin_amdgcn_wmma_f32_16x16x32_bf16(
      /*neg_a=*/false, a, /*neg_b=*/false, b,
      /*c_mod=*/(short)0, c, /*reuse_a=*/false, /*reuse_b=*/false);
}

// 16 contiguous input elements -> two bf16x8 (vectorized global path).
__device__ __forceinline__ void load16(const float* p, bf16x8& lo, bf16x8& hi) {
  const f32x4* q = (const f32x4*)p;
  f32x4 f0 = q[0], f1 = q[1], f2 = q[2], f3 = q[3];
#pragma unroll
  for (int j = 0; j < 4; ++j) {
    lo[j] = f2bf(f0[j]);
    lo[j + 4] = f2bf(f1[j]);
    hi[j] = f2bf(f2[j]);
    hi[j + 4] = f2bf(f3[j]);
  }
}
__device__ __forceinline__ void load16(const bf16_t* p, bf16x8& lo, bf16x8& hi) {
  const bf16x8* q = (const bf16x8*)p;
  lo = q[0];
  hi = q[1];
}

// ---------------------------------------------------------------------------
// Tiled GEMM: out[M,N] = X[M,K] @ W[K,N] + bias (+ resid).
// 256 threads / block = 8 waves; block tile 128x128; wave tile 32x64.
// A staged [m][k]; W staged transposed [n][k] via 4x4 register transpose so
// both operands feed WMMA with b128 fragment loads.
// ---------------------------------------------------------------------------
template <typename TIN, bool RESID>
__global__ __launch_bounds__(256) void gemm_bias_kernel(
    const TIN* __restrict__ X, const float* __restrict__ W,
    const float* __restrict__ bias, const float* __restrict__ resid,
    bf16_t* __restrict__ out_bf, float* __restrict__ out_f,
    int M, int N, int K) {
  constexpr int LDA = 40;  // 32 + 8 pad (80B rows: 16B-aligned, conflict-free)
  __shared__ __align__(16) bf16_t As[128 * LDA];
  __shared__ __align__(16) bf16_t BsT[128 * LDA];  // [n][k]

  const int tid = threadIdx.x;
  const int lane = tid & 31, w = tid >> 5;
  const int wr = w >> 1, wc = w & 1;  // 4x2 wave grid
  const int m0 = blockIdx.y * 128, n0 = blockIdx.x * 128;

  v8f acc[2][4];
#pragma unroll
  for (int i = 0; i < 2; ++i)
#pragma unroll
    for (int j = 0; j < 4; ++j)
#pragma unroll
      for (int r = 0; r < 8; ++r) acc[i][j][r] = 0.0f;

  const int nk = K / 32;
  for (int kt = 0; kt < nk; ++kt) {
    {  // A tile 128x32: each thread 16 contiguous elements of one row.
      int e0 = tid * 16;
      int r = e0 >> 5, k0 = e0 & 31;
      bf16x8 lo, hi;
      load16(&X[(size_t)(m0 + r) * K + kt * 32 + k0], lo, hi);
      *(bf16x8*)&As[r * LDA + k0] = lo;
      *(bf16x8*)&As[r * LDA + k0 + 8] = hi;
    }
    {  // W tile 32x128 -> BsT[n][k] via 4x4 register transpose.
      int n0t = (tid & 31) * 4;  // 0..124
      int k0t = (tid >> 5) * 4;  // 0..28
      f32x4 wrow[4];
#pragma unroll
      for (int kk = 0; kk < 4; ++kk)
        wrow[kk] = *(const f32x4*)&W[(size_t)(kt * 32 + k0t + kk) * N + n0 + n0t];
#pragma unroll
      for (int nn = 0; nn < 4; ++nn) {
        bf16x4 col;
#pragma unroll
        for (int kk = 0; kk < 4; ++kk) col[kk] = f2bf(wrow[kk][nn]);
        *(bf16x4*)&BsT[(n0t + nn) * LDA + k0t] = col;
      }
    }
    if (kt + 1 < nk) {  // gfx1250 global_prefetch_b8 for next k-tile
      __builtin_prefetch(&X[(size_t)(m0 + (tid & 127)) * K + (kt + 1) * 32], 0, 0);
      __builtin_prefetch(&W[(size_t)((kt + 1) * 32 + (tid & 31)) * N + n0], 0, 0);
    }
    __syncthreads();

    v16bf a0 = frag_row(&As[(wr * 32) * LDA], LDA);
    v16bf a1 = frag_row(&As[(wr * 32 + 16) * LDA], LDA);
#pragma unroll
    for (int nf = 0; nf < 4; ++nf) {
      v16bf bb = frag_row(&BsT[(wc * 64 + nf * 16) * LDA], LDA);
      acc[0][nf] = wmma_bf16(a0, bb, acc[0][nf]);
      acc[1][nf] = wmma_bf16(a1, bb, acc[1][nf]);
    }
    __syncthreads();
  }

  // Epilogue. C layout: lane = N (mod 16), vgpr r = M%8, lane>=16 -> M+8.
  const int hf = lane >> 4, nl = lane & 15;
#pragma unroll
  for (int mf = 0; mf < 2; ++mf)
#pragma unroll
    for (int nf = 0; nf < 4; ++nf)
#pragma unroll
      for (int r = 0; r < 8; ++r) {
        int m = m0 + wr * 32 + mf * 16 + r + hf * 8;
        int n = n0 + wc * 64 + nf * 16 + nl;
        float v = acc[mf][nf][r] + bias[n];
        if constexpr (RESID) {
          out_f[(size_t)m * N + n] = v + resid[(size_t)m * N + n];
        } else {
          out_bf[(size_t)m * N + n] = f2bf(v);
        }
      }
}

// ---------------------------------------------------------------------------
// Flash attention: one block per (b, h, 128-row q tile); each wave owns 16
// q rows. Streams 64-wide K/V chunks through LDS, online softmax, PV on WMMA.
// For S = Q K^T the B-operand layout [n=k_row][k=d] is K's NATURAL layout,
// so the K chunk is staged with gfx1250 async global->LDS b128 copies (no
// VGPR round-trip, overlaps WMMA). Only V needs a transposed stage for PV.
// ---------------------------------------------------------------------------
__global__ __launch_bounds__(256) void flash_attn_kernel(
    const bf16_t* __restrict__ Q, const bf16_t* __restrict__ Kb,
    const bf16_t* __restrict__ V, const float* __restrict__ amask,
    bf16_t* __restrict__ O) {
  constexpr int S = 2048, H = 1024, HD = 64, NH = 16;
  constexpr int LDK = 72;  // 64 + 8 pad (144B rows: 16B-aligned)
  __shared__ __align__(16) bf16_t Ks[64 * LDK];      // [k_row][d]
  __shared__ __align__(16) bf16_t VT[HD * LDK];      // [d][k_row]
  __shared__ __align__(16) bf16_t Ps[8 * 16 * LDK];  // per-wave P (16x64)

  const int tid = threadIdx.x, lane = tid & 31, w = tid >> 5;
  const int hf = lane >> 4, nl = lane & 15;
  const int qtiles = S / 128;
  const int qt = blockIdx.x % qtiles;
  const int bh = blockIdx.x / qtiles;
  const int b = bh / NH, h = bh % NH;
  const int q0 = qt * 128 + w * 16;  // this wave's q-row base

  // Resident Q A-fragments straight from global: 2x b128 per fragment.
  v16bf qa[2];
#pragma unroll
  for (int kk = 0; kk < 2; ++kk)
    qa[kk] = frag_row(&Q[(size_t)(b * S + q0) * H + h * HD + kk * 32], H);

  float mrow[8], lrow[8];
  v8f o[4];
#pragma unroll
  for (int r = 0; r < 8; ++r) { mrow[r] = -1e30f; lrow[r] = 0.0f; }
#pragma unroll
  for (int df = 0; df < 4; ++df)
#pragma unroll
    for (int r = 0; r < 8; ++r) o[df][r] = 0.0f;

  const float scale = 0.125f;  // 1/sqrt(64)

  for (int kc = 0; kc < S / 64; ++kc) {
    {  // K chunk, natural layout: async b128 global -> LDS (ASYNCcnt).
      int kr = tid >> 2;
      int d0 = (tid & 3) * 16;
      const bf16_t* gp = &Kb[(size_t)(b * S + kc * 64 + kr) * H + h * HD + d0];
      stage_b128(&Ks[kr * LDK + d0], gp);
      stage_b128(&Ks[kr * LDK + d0 + 8], gp + 8);
    }
    {  // V chunk transposed [d][k_row] via 4x4 register transpose.
      int kr0 = (tid & 15) * 4;
      int d0 = (tid >> 4) * 4;
      bf16x4 vrow[4];
#pragma unroll
      for (int i = 0; i < 4; ++i)
        vrow[i] =
            *(const bf16x4*)&V[(size_t)(b * S + kc * 64 + kr0 + i) * H + h * HD + d0];
#pragma unroll
      for (int j = 0; j < 4; ++j) {
        bf16x4 t;
#pragma unroll
        for (int i = 0; i < 4; ++i) t[i] = vrow[i][j];
        *(bf16x4*)&VT[(d0 + j) * LDK + kr0] = t;
      }
    }
    if (kc + 1 < S / 64) {
      size_t gi = (size_t)(b * S + (kc + 1) * 64 + (tid & 63)) * H + h * HD;
      __builtin_prefetch(&Kb[gi], 0, 0);
      __builtin_prefetch(&V[gi], 0, 0);
    }
    stage_wait();      // s_wait_asynccnt 0 (own async copies complete)
    __syncthreads();   // cross-wave visibility

    // S = Q K^T * scale + mask  (4 column fragments of 16 k_rows)
    v8f sf[4];
#pragma unroll
    for (int nf = 0; nf < 4; ++nf) {
      v8f c = {};
#pragma unroll
      for (int kk = 0; kk < 2; ++kk) {
        v16bf bb = frag_row(&Ks[(nf * 16) * LDK + kk * 32], LDK);
        c = wmma_bf16(qa[kk], bb, c);
      }
      float mv = amask[b * S + kc * 64 + nf * 16 + nl];
#pragma unroll
      for (int r = 0; r < 8; ++r) sf[nf][r] = c[r] * scale + mv;
    }

    // Row max over 64 columns: local over fragments, then 16-lane xor-reduce.
    float mx[8];
#pragma unroll
    for (int r = 0; r < 8; ++r) {
      mx[r] = fmaxf(fmaxf(sf[0][r], sf[1][r]), fmaxf(sf[2][r], sf[3][r]));
#pragma unroll
      for (int off = 1; off < 16; off <<= 1)
        mx[r] = fmaxf(mx[r], __shfl_xor(mx[r], off, 16));
    }
    // Online rescale.
#pragma unroll
    for (int r = 0; r < 8; ++r) {
      float mnew = fmaxf(mrow[r], mx[r]);
      float alpha = __expf(mrow[r] - mnew);
      mrow[r] = mnew;
      lrow[r] *= alpha;
#pragma unroll
      for (int df = 0; df < 4; ++df) o[df][r] *= alpha;
    }
    // P = exp(S - m); row sums; spill P to LDS (C layout -> [m][k] relayout).
    float rs[8];
#pragma unroll
    for (int r = 0; r < 8; ++r) rs[r] = 0.0f;
#pragma unroll
    for (int nf = 0; nf < 4; ++nf)
#pragma unroll
      for (int r = 0; r < 8; ++r) {
        float p = __expf(sf[nf][r] - mrow[r]);
        rs[r] += p;
        Ps[(w * 16 + r + hf * 8) * LDK + nf * 16 + nl] = f2bf(p);
      }
#pragma unroll
    for (int r = 0; r < 8; ++r) {
#pragma unroll
      for (int off = 1; off < 16; off <<= 1)
        rs[r] += __shfl_xor(rs[r], off, 16);
      lrow[r] += rs[r];
    }

    // O += P @ V   (contraction over the 64 k-rows of this chunk)
#pragma unroll
    for (int df = 0; df < 4; ++df)
#pragma unroll
      for (int kk = 0; kk < 2; ++kk) {
        v16bf aa = frag_row(&Ps[(w * 16) * LDK + kk * 32], LDK);
        v16bf bb = frag_row(&VT[(df * 16) * LDK + kk * 32], LDK);
        o[df] = wmma_bf16(aa, bb, o[df]);
      }
    __syncthreads();
  }

  // Normalize and write ctx in [B,S,H] (= [B,S,NH,HD]) layout, bf16.
#pragma unroll
  for (int df = 0; df < 4; ++df)
#pragma unroll
    for (int r = 0; r < 8; ++r) {
      int m = q0 + r + hf * 8;
      float val = o[df][r] / lrow[r];
      O[(size_t)(b * S + m) * H + h * HD + df * 16 + nl] = f2bf(val);
    }
}

// ---------------------------------------------------------------------------
// Row LayerNorm, eps=1e-12. One block (256 threads) per row of 1024.
// ---------------------------------------------------------------------------
__global__ __launch_bounds__(256) void layernorm_kernel(
    const float* __restrict__ X, const float* __restrict__ gamma,
    const float* __restrict__ beta, float* __restrict__ out, int Hdim) {
  __shared__ float red[256];
  const int row = blockIdx.x, tid = threadIdx.x;
  const float* xr = X + (size_t)row * Hdim;

  float v[4], s = 0.0f;
#pragma unroll
  for (int j = 0; j < 4; ++j) { v[j] = xr[tid + j * 256]; s += v[j]; }
  red[tid] = s;
  __syncthreads();
  for (int off = 128; off > 0; off >>= 1) {
    if (tid < off) red[tid] += red[tid + off];
    __syncthreads();
  }
  float mu = red[0] / (float)Hdim;
  __syncthreads();

  float sv = 0.0f;
#pragma unroll
  for (int j = 0; j < 4; ++j) { float d = v[j] - mu; sv += d * d; }
  red[tid] = sv;
  __syncthreads();
  for (int off = 128; off > 0; off >>= 1) {
    if (tid < off) red[tid] += red[tid + off];
    __syncthreads();
  }
  float rstd = rsqrtf(red[0] / (float)Hdim + 1e-12f);

#pragma unroll
  for (int j = 0; j < 4; ++j) {
    int c = tid + j * 256;
    out[(size_t)row * Hdim + c] = (v[j] - mu) * rstd * gamma[c] + beta[c];
  }
}

// ---------------------------------------------------------------------------
// Launch: QKV gemms -> flash attention -> O-proj (+resid) -> LayerNorm
// ---------------------------------------------------------------------------
extern "C" void kernel_launch(void* const* d_in, const int* in_sizes, int n_in,
                              void* d_out, int out_size, void* d_ws, size_t ws_size,
                              hipStream_t stream) {
  constexpr int B = 2, S = 2048, Hdim = 1024, NH = 16;
  constexpr int M = B * S;  // 4096

  const float* hidden = (const float*)d_in[0];
  const float* amask  = (const float*)d_in[1];
  const float* Wq = (const float*)d_in[2];
  const float* bq = (const float*)d_in[3];
  const float* Wk = (const float*)d_in[4];
  const float* bk = (const float*)d_in[5];
  const float* Wv = (const float*)d_in[6];
  const float* bv = (const float*)d_in[7];
  const float* Wo = (const float*)d_in[8];
  const float* bo = (const float*)d_in[9];
  const float* gamma = (const float*)d_in[10];
  const float* beta  = (const float*)d_in[11];

  char* ws = (char*)d_ws;
  const size_t bfbytes = (size_t)M * Hdim * sizeof(bf16_t);  // 8 MB each
  bf16_t* Qb  = (bf16_t*)(ws + 0 * bfbytes);
  bf16_t* Kbf = (bf16_t*)(ws + 1 * bfbytes);
  bf16_t* Vb  = (bf16_t*)(ws + 2 * bfbytes);
  bf16_t* Cb  = (bf16_t*)(ws + 3 * bfbytes);
  float*  Tmp = (float*)(ws + 4 * bfbytes);                  // 16 MB f32

  dim3 blk(256);
  dim3 gg(Hdim / 128, M / 128);  // (8, 32)

  gemm_bias_kernel<float, false><<<gg, blk, 0, stream>>>(
      hidden, Wq, bq, nullptr, Qb, nullptr, M, Hdim, Hdim);
  gemm_bias_kernel<float, false><<<gg, blk, 0, stream>>>(
      hidden, Wk, bk, nullptr, Kbf, nullptr, M, Hdim, Hdim);
  gemm_bias_kernel<float, false><<<gg, blk, 0, stream>>>(
      hidden, Wv, bv, nullptr, Vb, nullptr, M, Hdim, Hdim);

  flash_attn_kernel<<<dim3(B * NH * (S / 128)), blk, 0, stream>>>(
      Qb, Kbf, Vb, amask, Cb);

  gemm_bias_kernel<bf16_t, true><<<gg, blk, 0, stream>>>(
      Cb, Wo, bo, hidden, nullptr, Tmp, M, Hdim, Hdim);

  layernorm_kernel<<<dim3(M), blk, 0, stream>>>(Tmp, gamma, beta,
                                                (float*)d_out, Hdim);
}
